// ParallelMLPs_45775761441336
// MI455X (gfx1250) — matrix operand order repaired and checked by reading, not compiled, not run on hardware
//
#include <hip/hip_runtime.h>

#define NEXP   32
#define TOK    4096
#define DMODEL 256
#define HID    512

#define TM      128      // tokens per workgroup
#define THREADS 256      // 8 waves (wave32)

#define SX_LD 264        // x panel LDS row stride (elems): 256 + 8 pad (16B aligned)
#define SW_LD 264        // weight panel stride: 256 + 8 pad (matches TDM pad config)
#define SH_LD 520        // h panel stride: 512 + 8 pad

typedef __attribute__((ext_vector_type(16))) __bf16 v16bf;
typedef __attribute__((ext_vector_type(8)))  float  v8f;
typedef __attribute__((ext_vector_type(4)))  unsigned int u32x4;
typedef __attribute__((ext_vector_type(8)))  int i32x8;
typedef __attribute__((ext_vector_type(4)))  int i32x4;

__device__ __forceinline__ unsigned int pack2bf(float a, float b) {
  unsigned int ua = __float_as_uint(a);
  unsigned int ub = __float_as_uint(b);
  ua += 0x7FFFu + ((ua >> 16) & 1u);   // round-to-nearest-even
  ub += 0x7FFFu + ((ub >> 16) & 1u);
  return (ua >> 16) | (ub & 0xFFFF0000u);
}

// Stage a [rows x COLS] f32 panel (row stride g_ld) into LDS as bf16 (row stride s_ld).
template<int COLS>
__device__ __forceinline__ void stage_panel(const float* __restrict__ g, int g_ld,
                                            unsigned short* s, int s_ld, int rows) {
  const int tid = threadIdx.x;
  const int total = rows * COLS;
  for (int i = tid * 4; i < total; i += THREADS * 4) {
    const int r = i / COLS;
    const int c = i - r * COLS;
    const float4 f = *(const float4*)(g + (size_t)r * g_ld + c);
    uint2 p;
    p.x = pack2bf(f.x, f.y);
    p.y = pack2bf(f.z, f.w);
    *(uint2*)(s + r * s_ld + c) = p;
  }
}

// ---- Tensor Data Mover: DMA a [tile_d1 x tile_d0] bf16 tile (row-major,
// stride0 elems between rows) from global into LDS, padding each 512B row
// with 16B so the LDS row stride is SW_LD elems. D# per ISA 08_async_tensor §8.
__device__ __forceinline__ void tdm_load_panel(unsigned int lds_addr,
                                               const unsigned short* gptr,
                                               int tensor_d0, int tensor_d1,
                                               int tile_d0, int tile_d1, int stride0) {
  const unsigned long long ga = (unsigned long long)(size_t)gptr;
  u32x4 g0;
  g0.x = 1u;                                            // count=1, user descriptor
  g0.y = lds_addr;                                      // LDS byte address
  g0.z = (unsigned int)(ga & 0xFFFFFFFFu);              // global_addr[31:0]
  g0.w = (unsigned int)((ga >> 32) & 0x01FFFFFFu)       // global_addr[56:32]
       | (2u << 30);                                    // type=2 ("image")
  i32x8 g1;
  g1[0] = (1 << 16)                                     // data_size = 2 bytes
        | (1 << 20)                                     // pad_enable
        | (6 << 22)                                     // pad_interval: 128 DWORDs (512B row)
        | (3 << 25);                                    // pad_amount: 4 DWORDs (16B)
  g1[1] = (tensor_d0 & 0xFFFF) << 16;                   // tensor_dim0[15:0]
  g1[2] = ((tensor_d0 >> 16) & 0xFFFF)                  // tensor_dim0[31:16]
        | ((tensor_d1 & 0xFFFF) << 16);                 // tensor_dim1[15:0]
  g1[3] = ((tensor_d1 >> 16) & 0xFFFF)                  // tensor_dim1[31:16]
        | ((tile_d0 & 0xFFFF) << 16);                   // tile_dim0
  g1[4] = (tile_d1 & 0xFFFF);                           // tile_dim1 (tile_dim2=0 -> 2D)
  g1[5] = stride0;                                      // tensor_dim0_stride[31:0]
  g1[6] = 0;
  g1[7] = 0;
  const i32x4 z4 = {0, 0, 0, 0};
  const i32x8 z8 = {0, 0, 0, 0, 0, 0, 0, 0};
  __builtin_amdgcn_tensor_load_to_lds(g0, g1, z4, z4, z8, 0);
}

// Assemble a 16-element bf16 WMMA operand from two 16B LDS chunks.
__device__ __forceinline__ v16bf load_mat16(const unsigned short* p0, const unsigned short* p1) {
  union { uint4 q[2]; v16bf v; } u;
  u.q[0] = *(const uint4*)p0;
  u.q[1] = *(const uint4*)p1;
  return u.v;
}

template<bool USE_TDM>
__global__ __launch_bounds__(THREADS, 1)
void parallel_mlp_kernel(const float* __restrict__ x,  const float* __restrict__ W1,
                         const float* __restrict__ b1, const float* __restrict__ W2,
                         const float* __restrict__ b2,
                         const unsigned short* __restrict__ w1bf,
                         const unsigned short* __restrict__ w2bf,
                         float* __restrict__ out) {
  __shared__ __align__(16) unsigned short SX[TM * SX_LD];  //  66 KB: x tile bf16
  __shared__ __align__(16) unsigned short SH[TM * SH_LD];  // 130 KB: silu(h) bf16
  __shared__ __align__(16) unsigned short SW[TM * SW_LD];  //  66 KB: streaming W panel

  const int e  = blockIdx.x / (TOK / TM);
  const int tb = blockIdx.x % (TOK / TM);
  const int t0 = tb * TM;

  const int tid   = threadIdx.x;
  const int wave  = tid >> 5;        // wave32
  const int lane  = tid & 31;
  const int lhalf = lane >> 4;       // 0: lanes 0-15, 1: lanes 16-31
  const int l16   = lane & 15;
  const int Mbase = wave * 16;       // each wave owns 16 token rows

  const float* xg  = x  + ((size_t)e * TOK + t0) * DMODEL;
  const float* w1g = W1 + (size_t)e * HID * DMODEL;
  const float* w2g = W2 + (size_t)e * DMODEL * HID;
  const unsigned short* w1b = w1bf + (size_t)e * HID * DMODEL;
  const unsigned short* w2b = w2bf + (size_t)e * DMODEL * HID;
  (void)w1g; (void)w2g; (void)w1b; (void)w2b;

  // Stage x tile [TM][DMODEL] -> bf16 LDS (single-use data: convert in-kernel)
  stage_panel<DMODEL>(xg, DMODEL, SX, SX_LD, TM);

  // ---------- Phase 1: h = silu(x @ W1^T + b1), kept in LDS as bf16 ----------
  for (int hblk = 0; hblk < HID / TM; ++hblk) {        // 4 blocks of 128 H-cols
    const int h0 = hblk * TM;
    __syncthreads();                                   // prior panel fully consumed
    if constexpr (USE_TDM) {
      if (wave == 0) {                                 // one wave drives the TDM
        tdm_load_panel((unsigned int)(size_t)SW, w1b + (size_t)h0 * DMODEL,
                       DMODEL, HID, DMODEL, TM, DMODEL);
        __builtin_amdgcn_s_wait_tensorcnt(0);
      }
    } else {
      stage_panel<DMODEL>(w1g + (size_t)h0 * DMODEL, DMODEL, SW, SW_LD, TM);
    }
    __syncthreads();                                   // panel visible to all waves

    v8f acc[8];
#pragma unroll
    for (int nt = 0; nt < 8; ++nt) acc[nt] = (v8f)0.f;

#pragma unroll
    for (int kk = 0; kk < DMODEL; kk += 32) {
      const unsigned short* arow = SX + (Mbase + l16) * SX_LD;
      const v16bf a = load_mat16(arow + kk + lhalf * 8,
                                 arow + kk + 16 + lhalf * 8);
#pragma unroll
      for (int nt = 0; nt < 8; ++nt) {
        const unsigned short* brow = SW + (nt * 16 + l16) * SW_LD + kk + lhalf * 8;
        const v16bf b = load_mat16(brow, brow + 16);
        acc[nt] = __builtin_amdgcn_wmma_f32_16x16x32_bf16(
            false, a, false, b, (short)0, acc[nt], false, false);
      }
    }

    // Epilogue: bias + SiLU, write bf16 into SH (each wave writes only its rows)
#pragma unroll
    for (int nt = 0; nt < 8; ++nt) {
      const int hcol = h0 + nt * 16 + l16;
      const float bias = b1[e * HID + hcol];
#pragma unroll
      for (int v = 0; v < 8; ++v) {
        const int row = Mbase + v + lhalf * 8;
        float hval = acc[nt][v] + bias;
        hval = hval / (1.f + __expf(-hval));           // SiLU
        unsigned int u = __float_as_uint(hval);
        u += 0x7FFFu + ((u >> 16) & 1u);
        SH[row * SH_LD + hcol] = (unsigned short)(u >> 16);
      }
    }
  }

  // ---------- Phase 2: out = h @ W2^T + b2 + x ----------
  for (int dblk = 0; dblk < DMODEL / TM; ++dblk) {     // 2 blocks of 128 D-cols
    const int d0 = dblk * TM;
    v8f acc[8];
#pragma unroll
    for (int nt = 0; nt < 8; ++nt) acc[nt] = (v8f)0.f;

    for (int hc = 0; hc < HID / 256; ++hc) {           // 2 K-panels of 256
      const int h0 = hc * 256;
      __syncthreads();
      if constexpr (USE_TDM) {
        if (wave == 0) {
          tdm_load_panel((unsigned int)(size_t)SW, w2b + (size_t)d0 * HID + h0,
                         HID, DMODEL, 256, TM, HID);
          __builtin_amdgcn_s_wait_tensorcnt(0);
        }
      } else {
        stage_panel<256>(w2g + (size_t)d0 * HID + h0, HID, SW, SW_LD, TM);
      }
      __syncthreads();

#pragma unroll
      for (int kk = 0; kk < 256; kk += 32) {
        const unsigned short* arow = SH + (Mbase + l16) * SH_LD + h0;
        const v16bf a = load_mat16(arow + kk + lhalf * 8,
                                   arow + kk + 16 + lhalf * 8);
#pragma unroll
        for (int nt = 0; nt < 8; ++nt) {
          const unsigned short* brow = SW + (nt * 16 + l16) * SW_LD + kk + lhalf * 8;
          const v16bf b = load_mat16(brow, brow + 16);
          acc[nt] = __builtin_amdgcn_wmma_f32_16x16x32_bf16(
              false, a, false, b, (short)0, acc[nt], false, false);
        }
      }
    }

    // Epilogue: + b2 + residual x, store f32
#pragma unroll
    for (int nt = 0; nt < 8; ++nt) {
      const int dcol = d0 + nt * 16 + l16;
      const float bias = b2[e * DMODEL + dcol];
#pragma unroll
      for (int v = 0; v < 8; ++v) {
        const int row = Mbase + v + lhalf * 8;
        const size_t gi = ((size_t)e * TOK + t0 + row) * DMODEL + dcol;
        out[gi] = acc[nt][v] + bias + x[gi];
      }
    }
  }
}

// One-shot f32 -> bf16 weight conversion into workspace (W1 then W2, flat).
__global__ __launch_bounds__(256)
void convert_weights(const float* __restrict__ W1, const float* __restrict__ W2,
                     unsigned short* __restrict__ ws) {
  const long long n1 = (long long)NEXP * HID * DMODEL;       // 4,194,304
  const long long i  = ((long long)blockIdx.x * blockDim.x + threadIdx.x) * 4;
  if (i >= 2 * n1) return;
  const float* src = (i < n1) ? (W1 + i) : (W2 + (i - n1));
  const float4 f = *(const float4*)src;
  uint2 p;
  p.x = pack2bf(f.x, f.y);
  p.y = pack2bf(f.z, f.w);
  *(uint2*)(ws + i) = p;
}

extern "C" void kernel_launch(void* const* d_in, const int* in_sizes, int n_in,
                              void* d_out, int out_size, void* d_ws, size_t ws_size,
                              hipStream_t stream) {
  const float* x  = (const float*)d_in[0];
  const float* W1 = (const float*)d_in[1];
  const float* b1 = (const float*)d_in[2];
  const float* W2 = (const float*)d_in[3];
  const float* b2 = (const float*)d_in[4];
  float* out = (float*)d_out;

  const dim3 grid(NEXP * (TOK / TM));   // 32 experts * 32 token tiles = 1024 WGs
  const size_t welems = 2ull * NEXP * HID * DMODEL;          // W1 + W2 elements
  const size_t need   = welems * sizeof(unsigned short);     // 16.8 MB bf16

  if (ws_size >= need && d_ws != nullptr) {
    unsigned short* wbf = (unsigned short*)d_ws;
    const int cblocks = (int)((welems / 4 + 255) / 256);     // 8192 blocks
    convert_weights<<<cblocks, 256, 0, stream>>>(W1, W2, wbf);
    const unsigned short* w1bf = wbf;
    const unsigned short* w2bf = wbf + (size_t)NEXP * HID * DMODEL;
    parallel_mlp_kernel<true><<<grid, THREADS, 0, stream>>>(
        x, W1, b1, W2, b2, w1bf, w2bf, out);
  } else {
    parallel_mlp_kernel<false><<<grid, THREADS, 0, stream>>>(
        x, W1, b1, W2, b2, nullptr, nullptr, out);
  }
}
